// Net_61211873903126
// MI455X (gfx1250) — matrix operand order repaired and checked by reading, not compiled, not run on hardware
//
#include <hip/hip_runtime.h>
#include <math.h>

// ---------------------------------------------------------------------------
// Shapes / constants from the reference
// ---------------------------------------------------------------------------
#define B_SEQ   512     // scan length (the "B" axis is scanned over!)
#define IN_D    64
#define H_D     512
#define OUT_D   12
#define T_STEPS 31
#define S0_CUT  8
#define NWG     32      // persistent workgroups; each owns 16 h-elements
#define WG_THREADS 128  // 4 waves (wave32) = one wave per LSTM gate

typedef _Float16 v16h __attribute__((ext_vector_type(16)));
typedef _Float16 v4h  __attribute__((ext_vector_type(4)));
typedef float    v8f  __attribute__((ext_vector_type(8)));

// ---------------------------------------------------------------------------
// Workspace layout (bytes).  Packed f16 weights (A-fragment order) dominate:
//   3 matrices x 2048x512 halfs = 6 MB  -> permanently L2-resident on MI455X
//   (and register-resident inside the persistent kernel).
// ---------------------------------------------------------------------------
#define PW_MAT    (2048ull*512ull)                 // halfs per matrix
#define PW_OFF    0ull                             // 6,291,456 B
#define XB_OFF    6291456ull                       // xb0: 2048 f32 (b0 + Wih0@1)
#define HB0_OFF   6299648ull                       // layer0 h (f16, 512)
#define HB1_OFF   6300672ull                       // layer1 h (f16, 512)
#define YS0_OFF   6301696ull                       // ys0 f16 [512][512]
#define YS1F_OFF  6825984ull                       // ys1 f32 [512][512]
#define SCALE_OFF 7874560ull                       // BN scale f32[512]
#define SHIFT_OFF 7876608ull                       // BN shift f32[512]
#define BAR_OFF   7878656ull                       // grid barrier {count, gen}

// ---------------------------------------------------------------------------
// One-time weight pack: f32 row-major -> f16 WMMA-A-fragment order.
// Packed index p (per matrix): ((((w*4+q)*16+kc)*32+lane)*16+t)
//   lane<16  holds row (w*16 + lane),    K = {0..15} of the 32-chunk
//   lane>=16 holds row (w*16 + lane-16), K = {16..31}
// ---------------------------------------------------------------------------
__global__ void pack_weights_kernel(const float* __restrict__ Whh0,
                                    const float* __restrict__ Wih1,
                                    const float* __restrict__ Whh1,
                                    _Float16* __restrict__ pW) {
  int p = blockIdx.x * blockDim.x + threadIdx.x;
  if (p >= 3 * 1048576) return;
  int mat  = p >> 20;
  int r    = p & 1048575;
  int t    = r & 15;
  int lane = (r >> 4) & 31;
  int kc   = (r >> 9) & 15;
  int q    = (r >> 13) & 3;
  int w    = r >> 15;
  int hi   = lane >> 4;
  int m    = lane & 15;
  int row  = q * 512 + w * 16 + m;
  int koff = t + hi * 16;
  int col  = kc * 32 + koff;
  const float* src = (mat == 0) ? Whh0 : ((mat == 1) ? Wih1 : Whh1);
  pW[p] = (_Float16)src[row * 512 + col];
}

// xb0 = b0 + Wih0 @ ones(64)  (x is all-ones -> constant for the whole run),
// plus h-state init and barrier reset (ws is poisoned by the harness).
__global__ void init_kernel(const float* __restrict__ Wih0,
                            const float* __restrict__ b0,
                            const float* __restrict__ h0,
                            float* __restrict__ xb0,
                            _Float16* __restrict__ hb0,
                            _Float16* __restrict__ hb1,
                            unsigned* __restrict__ bar) {
  int r = blockIdx.x * blockDim.x + threadIdx.x;
  if (r < 2048) {
    float s = b0[r];
    #pragma unroll 8
    for (int c = 0; c < IN_D; ++c) s += Wih0[r * IN_D + c];
    xb0[r] = s;
  }
  if (r < 512) {
    hb0[r] = (_Float16)h0[r];
    hb1[r] = (_Float16)h0[512 + r];
  }
  if (r == 0) { bar[0] = 0u; bar[1] = 0u; }
}

// ---------------------------------------------------------------------------
// Device helpers
// ---------------------------------------------------------------------------
__device__ __forceinline__ float sigf(float x) { return 1.f / (1.f + __expf(-x)); }

// Counter-based Gumbel noise (stand-in for jax.random.categorical's threefry;
// deterministic pure function of (iter, row, class)).
__device__ __forceinline__ float gumbelh(unsigned i, unsigned b, unsigned o) {
  unsigned s = (i + 1u) * 0x9E3779B1u ^ (b + 0x7F4A7C15u) * 0x85EBCA77u ^
               (o + 1u) * 0xC2B2AE3Du;
  s ^= s >> 15; s *= 0x2C1B3C6Du; s ^= s >> 12; s *= 0x297A2D39u; s ^= s >> 15;
  float u = (float)(s >> 8) * (1.0f / 16777216.0f) + 2.98e-8f;
  return -__logf(-__logf(u));
}

// Accumulate y[16] += Wslice(16x512) @ vec(512) with 16 chained
// v_wmma_f32_16x16x32_f16.  A-tiles are register-resident (preloaded once).
// B-fragments use an explicit 2-deep software pipeline: the WMMA for chunk kc
// consumes bcur (loads issued one stage earlier) while bnext's ds_loads are
// already in flight, so the wait before each WMMA is dscnt<=2, not a full
// drain, and LDS latency overlaps the matrix pipe — critical at 1 wave/SIMD.
// B replicates the vector across all 16 N columns, so every D column holds
// the dot product and extraction is independent of the N-lane mapping.
__device__ __forceinline__ v8f wmma_chain16(const v16h* __restrict__ a,
                                            const _Float16* fsrc, int lane, v8f acc) {
  const int hi16 = (lane >> 4) & 1;
  const _Float16* base = fsrc + hi16 * 16;
  v16h bcur = *(const v16h*)base;                    // stage-0 loads
  #pragma unroll
  for (int kc = 0; kc < 16; ++kc) {
    v16h bnext = (kc < 15) ? *(const v16h*)(base + (kc + 1) * 32) : bcur;
    acc = __builtin_amdgcn_wmma_f32_16x16x32_f16(
        /*neg_a=*/false, a[kc], /*neg_b=*/false, bcur,
        /*c_mod=*/(short)0, acc, /*reuse_a=*/false, /*reuse_b=*/false);
    bcur = bnext;
  }
  return acc;
}

// Device-scope barrier across the NWG persistent workgroups.  (On a cluster
// dispatch the HW path would be s_barrier_signal/-3; a normal dispatch has
// ClusterID==0 so those are NOPs -> atomic generation barrier through L2.)
__device__ __forceinline__ void grid_barrier(unsigned* cnt, unsigned* genp, int tid) {
  __syncthreads();
  if (tid == 0) {
    volatile unsigned* gen = genp;
    __threadfence();
    unsigned old = *gen;
    if (atomicAdd(cnt, 1u) == (unsigned)(NWG - 1)) {
      *cnt = 0u;
      __threadfence();
      atomicAdd(genp, 1u);
    } else {
      while (*gen == old) { __builtin_amdgcn_s_sleep(2); }
    }
  }
  __syncthreads();
}

// ---------------------------------------------------------------------------
// Persistent kernel: 32 WGs x 128 threads.  WG w owns h-elements [16w,16w+16);
// wave q (of 4) owns gate q (torch order i,f,g,o) and keeps its 48 A-tiles
// (3 matrices x 16 k-chunks x 8 VGPRs) register-resident for the entire run
// via the gfx1250 1024-VGPR file.  Per LSTM step: pipelined B-frag feed from
// LDS, WMMA chain, pointwise cell update by 16 threads, then the 512-wide h
// vector is exchanged through L2 under a device barrier.
// ---------------------------------------------------------------------------
__global__ void __launch_bounds__(WG_THREADS, 1)
lstm_persistent_kernel(const _Float16* __restrict__ pW,
                       const float* __restrict__ xb0,
                       _Float16* __restrict__ hb0,
                       _Float16* __restrict__ hb1,
                       _Float16* __restrict__ ys0,
                       float* __restrict__ ys1f,
                       float* __restrict__ scl,
                       float* __restrict__ shf,
                       unsigned* __restrict__ bar,
                       const float* __restrict__ b1,
                       const float* __restrict__ gma,
                       const float* __restrict__ bta,
                       const float* __restrict__ Wout,
                       const float* __restrict__ bout,
                       const float* __restrict__ c0,
                       int* __restrict__ out_act,
                       float* __restrict__ out_lp) {
  __shared__ _Float16 hsh[512];
  __shared__ _Float16 xsh[512];
  __shared__ float    gsh[64];
  __shared__ float    csh0[16], csh1[16];
  __shared__ float    wsh[OUT_D * 512];
  __shared__ float    ssh[512], shsh[512];

  const int tid  = threadIdx.x;
  const int w    = blockIdx.x;
  const int lane = tid & 31;
  const int wv   = tid >> 5;  // gate id 0..3

  for (int idx = tid; idx < OUT_D * 512; idx += WG_THREADS) wsh[idx] = Wout[idx];
  if (tid < 16) {
    csh0[tid] = c0[w * 16 + tid];
    csh1[tid] = c0[512 + w * 16 + tid];
  }
  __syncthreads();

  const size_t slice = (size_t)w * 32768 + (size_t)wv * 8192;
  const _Float16* Whh0s = pW + 0 * PW_MAT + slice;
  const _Float16* Wih1s = pW + 1 * PW_MAT + slice;
  const _Float16* Whh1s = pW + 2 * PW_MAT + slice;

  // Explicitly preload all 48 A-tiles (384 VGPRs/lane) — weights never touch
  // memory again for the remaining 15,872 recurrence steps.
  v16h a0[16], a1[16], a2[16];
  #pragma unroll
  for (int kc = 0; kc < 16; ++kc) {
    const size_t o = (size_t)kc * 512 + (size_t)lane * 16;
    a0[kc] = *(const v16h*)(Whh0s + o);
    a1[kc] = *(const v16h*)(Wih1s + o);
    a2[kc] = *(const v16h*)(Whh1s + o);
  }

  unsigned* cnt = bar;
  unsigned* gen = bar + 1;

  for (int it = 0; it < T_STEPS; ++it) {
    // ---------------- layer 0: 512 sequential cell steps ----------------
    for (int t = 0; t < B_SEQ; ++t) {
      ((v4h*)hsh)[tid] = ((const v4h*)hb0)[tid];   // 8B/lane: b64 load + ds b64
      __syncthreads();

      v8f acc = {0.f, 0.f, 0.f, 0.f, 0.f, 0.f, 0.f, 0.f};
      acc = wmma_chain16(a0, hsh, lane, acc);

      if (lane == 0) {
        #pragma unroll
        for (int r = 0; r < 8; ++r) gsh[wv * 16 + r] = acc[r];
      }
      if (lane == 16) {
        #pragma unroll
        for (int r = 0; r < 8; ++r) gsh[wv * 16 + 8 + r] = acc[r];
      }
      __syncthreads();

      if (tid < 16) {
        const int col = w * 16 + tid;
        float gi = gsh[tid]      + xb0[col];
        float gf = gsh[16 + tid] + xb0[512 + col];
        float gc = gsh[32 + tid] + xb0[1024 + col];
        float go = gsh[48 + tid] + xb0[1536 + col];
        float cn = sigf(gf) * csh0[tid] + sigf(gi) * tanhf(gc);
        float hn = sigf(go) * tanhf(cn);
        csh0[tid] = cn;
        _Float16 hh = (_Float16)hn;
        hb0[col] = hh;
        ys0[(size_t)t * 512 + col] = hh;
      }
      grid_barrier(cnt, gen, tid);
    }

    // ---------------- layer 1: 512 sequential cell steps ----------------
    for (int t = 0; t < B_SEQ; ++t) {
      ((v4h*)hsh)[tid] = ((const v4h*)hb1)[tid];
      ((v4h*)xsh)[tid] = ((const v4h*)(ys0 + (size_t)t * 512))[tid];
      __syncthreads();

      v8f acc = {0.f, 0.f, 0.f, 0.f, 0.f, 0.f, 0.f, 0.f};
      acc = wmma_chain16(a1, xsh, lane, acc);   // Wih1 @ ys0[t]
      acc = wmma_chain16(a2, hsh, lane, acc);   // Whh1 @ h1

      if (lane == 0) {
        #pragma unroll
        for (int r = 0; r < 8; ++r) gsh[wv * 16 + r] = acc[r];
      }
      if (lane == 16) {
        #pragma unroll
        for (int r = 0; r < 8; ++r) gsh[wv * 16 + 8 + r] = acc[r];
      }
      __syncthreads();

      if (tid < 16) {
        const int col = w * 16 + tid;
        float gi = gsh[tid]      + b1[col];
        float gf = gsh[16 + tid] + b1[512 + col];
        float gc = gsh[32 + tid] + b1[1024 + col];
        float go = gsh[48 + tid] + b1[1536 + col];
        float cn = sigf(gf) * csh1[tid] + sigf(gi) * tanhf(gc);
        float hn = sigf(go) * tanhf(cn);
        csh1[tid] = cn;
        hb1[col] = (_Float16)hn;
        ys1f[(size_t)t * 512 + col] = hn;
      }
      grid_barrier(cnt, gen, tid);
    }

    // ---------------- head: BN stats -> gaussian -> logits -> sample -----
    const int gt = w * WG_THREADS + tid;  // 0..4095
    if (gt < 512) {                       // per-feature BN (training mode)
      float s = 0.f, s2 = 0.f;
      for (int bb = 0; bb < 512; ++bb) {
        float v = ys1f[(size_t)bb * 512 + gt];
        s += v;
        s2 = fmaf(v, v, s2);
      }
      float mu  = s * (1.f / 512.f);
      float var = s2 * (1.f / 512.f) - mu * mu;
      float is  = rsqrtf(var + 1e-5f);
      float sc  = gma[gt] * is;
      scl[gt] = sc;
      shf[gt] = bta[gt] - mu * sc;
    }
    grid_barrier(cnt, gen, tid);

    for (int idx = tid; idx < 512; idx += WG_THREADS) {
      ssh[idx]  = scl[idx];
      shsh[idx] = shf[idx];
    }
    __syncthreads();

    if (gt < 512) {                       // one thread per output row
      const int bb = gt;
      float lg[OUT_D];
      #pragma unroll
      for (int o = 0; o < OUT_D; ++o) lg[o] = bout[o];
      for (int k = 0; k < 512; ++k) {
        float v  = ys1f[(size_t)bb * 512 + k];
        float yn = fmaf(v, ssh[k], shsh[k]);
        float gv = __expf(-yn * yn);      // GaussianActivation
        #pragma unroll
        for (int o = 0; o < OUT_D; ++o) lg[o] = fmaf(gv, wsh[o * 512 + k], lg[o]);
      }
      const int lo = (it < 15) ? 0 : S0_CUT;  // i < HEAD -> full categorical
      float best = -1e30f; int act = lo;
      for (int o = lo; o < OUT_D; ++o) {
        float z = lg[o] + gumbelh((unsigned)it, (unsigned)bb, (unsigned)o);
        if (z > best) { best = z; act = o; }
      }
      float mx = lg[0];
      #pragma unroll
      for (int o = 1; o < OUT_D; ++o) mx = fmaxf(mx, lg[o]);
      float se = 0.f;
      #pragma unroll
      for (int o = 0; o < OUT_D; ++o) se += __expf(lg[o] - mx);
      out_act[it * 512 + bb] = act;                       // (T,B) int32
      out_lp[it * 512 + bb]  = lg[act] - mx - __logf(se); // (T,B) f32, full softmax
    }
    grid_barrier(cnt, gen, tid);
  }
}

// ---------------------------------------------------------------------------
// Host entry
// ---------------------------------------------------------------------------
extern "C" void kernel_launch(void* const* d_in, const int* in_sizes, int n_in,
                              void* d_out, int out_size, void* d_ws, size_t ws_size,
                              hipStream_t stream) {
  (void)in_sizes; (void)n_in; (void)out_size; (void)ws_size;
  const float* Wih0 = (const float*)d_in[0];
  const float* Whh0 = (const float*)d_in[1];
  const float* b0   = (const float*)d_in[2];
  const float* Wih1 = (const float*)d_in[3];
  const float* Whh1 = (const float*)d_in[4];
  const float* b1   = (const float*)d_in[5];
  const float* gma  = (const float*)d_in[6];
  const float* bta  = (const float*)d_in[7];
  const float* Wout = (const float*)d_in[8];
  const float* bout = (const float*)d_in[9];
  const float* h0   = (const float*)d_in[10];
  const float* c0   = (const float*)d_in[11];

  char* ws = (char*)d_ws;
  _Float16* pW  = (_Float16*)(ws + PW_OFF);
  float*    xb0 = (float*)(ws + XB_OFF);
  _Float16* hb0 = (_Float16*)(ws + HB0_OFF);
  _Float16* hb1 = (_Float16*)(ws + HB1_OFF);
  _Float16* ys0 = (_Float16*)(ws + YS0_OFF);
  float*   ys1f = (float*)(ws + YS1F_OFF);
  float*    scl = (float*)(ws + SCALE_OFF);
  float*    shf = (float*)(ws + SHIFT_OFF);
  unsigned* bar = (unsigned*)(ws + BAR_OFF);

  int*   out_act = (int*)d_out;                     // first T*B: actions
  float* out_lp  = (float*)d_out + T_STEPS * B_SEQ; // next T*B: log-probs

  pack_weights_kernel<<<(3 * 1048576) / 256, 256, 0, stream>>>(Whh0, Wih1, Whh1, pW);
  init_kernel<<<8, 256, 0, stream>>>(Wih0, b0, h0, xb0, hb0, hb1, bar);
  lstm_persistent_kernel<<<NWG, WG_THREADS, 0, stream>>>(
      pW, xb0, hb0, hb1, ys0, ys1f, scl, shf, bar,
      b1, gma, bta, Wout, bout, c0, out_act, out_lp);
}
